// SlidingWindowAttention_65687229825394
// MI455X (gfx1250) — matrix operand order repaired and checked by reading, not compiled, not run on hardware
//
#include <hip/hip_runtime.h>

typedef _Float16 v16h __attribute__((ext_vector_type(16)));
typedef _Float16 v8h  __attribute__((ext_vector_type(8)));
typedef float    v8f  __attribute__((ext_vector_type(8)));

#define T_SEQ   2048
#define D_MOD   1024
#define NHEAD   16
#define DHEAD   64
#define HALFWIN 128

__device__ __forceinline__ v8f wmma_f16(v16h a, v16h b, v8f c) {
    return __builtin_amdgcn_wmma_f32_16x16x32_f16(
        /*neg_a=*/false, a, /*neg_b=*/false, b,
        /*c_mod=*/(short)0, c, /*reuse_a=*/false, /*reuse_b=*/false);
}

// Build a 16-half fragment from two contiguous 16B f16 chunks.
__device__ __forceinline__ v16h ld16(const _Float16* p0, const _Float16* p1) {
    v8h lo = *(const v8h*)p0;
    v8h hi = *(const v8h*)p1;
    v16h r;
#pragma unroll
    for (int i = 0; i < 8; ++i) { r[i] = lo[i]; r[i + 8] = hi[i]; }
    return r;
}

// Build a 16-half fragment from two runs of 8 contiguous f32 in LDS.
__device__ __forceinline__ v16h cvt16(const float* p0, const float* p1) {
    float4 a = *(const float4*)p0;
    float4 b = *(const float4*)(p0 + 4);
    float4 c = *(const float4*)p1;
    float4 d = *(const float4*)(p1 + 4);
    v16h r;
    r[0]  = (_Float16)a.x; r[1]  = (_Float16)a.y; r[2]  = (_Float16)a.z; r[3]  = (_Float16)a.w;
    r[4]  = (_Float16)b.x; r[5]  = (_Float16)b.y; r[6]  = (_Float16)b.z; r[7]  = (_Float16)b.w;
    r[8]  = (_Float16)c.x; r[9]  = (_Float16)c.y; r[10] = (_Float16)c.z; r[11] = (_Float16)c.w;
    r[12] = (_Float16)d.x; r[13] = (_Float16)d.y; r[14] = (_Float16)d.z; r[15] = (_Float16)d.w;
    return r;
}

// Issue one per-lane async DMA of 16 bytes global -> LDS (ASYNCcnt-tracked).
__device__ __forceinline__ void async_b128(uint32_t lds_off, const void* gptr) {
    unsigned long long ga = (unsigned long long)(uintptr_t)gptr;
    asm volatile("global_load_async_to_lds_b128 %0, %1, off"
                 :: "v"(lds_off), "v"(ga) : "memory");
}
__device__ __forceinline__ void wait_async0() {
    asm volatile("s_wait_asynccnt 0x0" ::: "memory");
}

// ---------------------------------------------------------------------------
// Kernel 1: fused QKV projection.  y[t,o] = sum_d x[t,d] * W[o,d]
// Async double-buffered f32 LDS staging; f32->f16 conversion at fragment read.
// Q scaled by 1/sqrt(dh), stored [bh][T][dh] f16.  K same (unscaled).
// V stored transposed [bh][dh][T] f16 so attention P*V B-fragments are
// contiguous.
// ---------------------------------------------------------------------------
__global__ __launch_bounds__(256) void qkv_gemm_kernel(
    const float* __restrict__ x, const float* __restrict__ Wq,
    const float* __restrict__ Wk, const float* __restrict__ Wv,
    _Float16* __restrict__ Qb, _Float16* __restrict__ Kb,
    _Float16* __restrict__ Vt) {
    __shared__ float As[2][64 * 44];  // 64 rows x 32 k (f32), stride 44
    __shared__ float Bs[2][64 * 44];

    const int m0 = blockIdx.x * 64;
    const int n0 = blockIdx.y * 64;          // 0..3071
    const int mat = n0 >> 10;                // 0=Q 1=K 2=V
    const float* __restrict__ W = (mat == 0) ? Wq : (mat == 1) ? Wk : Wv;
    const int wrow0 = n0 & 1023;

    const int tid  = threadIdx.x;
    const int lane = tid & 31;
    const int wv   = tid >> 5;
    const int wm   = wv & 3;                 // wave row (4)
    const int wn   = wv >> 2;                // wave col (2)
    const int lm   = lane & 15;
    const int hb   = (lane < 16) ? 0 : 8;    // C-layout row offset
    const int kb   = (lane < 16) ? 0 : 8;    // A-fragment K base
    const int ks   = (lane < 16) ? 0 : 16;   // B-fragment K base

    const int arow = tid >> 2;               // 0..63 staging row
    const int aq   = tid & 3;                // 0..3 k-quad (8 floats)

    const float* xg = x + (size_t)(m0 + arow) * D_MOD + aq * 8;
    const float* wg = W + (size_t)(wrow0 + arow) * D_MOD + aq * 8;
    const uint32_t ldsA0 = (uint32_t)(size_t)&As[0][arow * 44 + aq * 8];
    const uint32_t ldsA1 = (uint32_t)(size_t)&As[1][arow * 44 + aq * 8];
    const uint32_t ldsB0 = (uint32_t)(size_t)&Bs[0][arow * 44 + aq * 8];
    const uint32_t ldsB1 = (uint32_t)(size_t)&Bs[1][arow * 44 + aq * 8];

    v8f acc[2];
#pragma unroll
    for (int nt = 0; nt < 2; ++nt) acc[nt] = v8f{};

    // prologue: fill buffer 0 for k=0
    async_b128(ldsA0, xg);      async_b128(ldsA0 + 16, xg + 4);
    async_b128(ldsB0, wg);      async_b128(ldsB0 + 16, wg + 4);

    for (int it = 0; it < 32; ++it) {
        const int cur = it & 1;
        wait_async0();           // current buffer's DMA complete (this wave)
        __syncthreads();         // ... and every other wave's
        if (it + 1 < 32) {       // kick off next tile into the other buffer
            const int kn = (it + 1) * 32;
            const uint32_t la = cur ? ldsA0 : ldsA1;
            const uint32_t lb = cur ? ldsB0 : ldsB1;
            async_b128(la, xg + kn);      async_b128(la + 16, xg + kn + 4);
            async_b128(lb, wg + kn);      async_b128(lb + 16, wg + kn + 4);
        }
        const int mloc = wm * 16 + lm;
        v16h af = cvt16(&As[cur][mloc * 44 + kb], &As[cur][mloc * 44 + 16 + kb]);
#pragma unroll
        for (int nt = 0; nt < 2; ++nt) {
            const int nloc = wn * 32 + nt * 16 + lm;
            v16h bf = cvt16(&Bs[cur][nloc * 44 + kb], &Bs[cur][nloc * 44 + 16 + kb]);
            acc[nt] = wmma_f16(af, bf, acc[nt]);
        }
    }

    // scatter store into attention-friendly layouts
#pragma unroll
    for (int r = 0; r < 8; ++r) {
        const int gr = m0 + wm * 16 + r + hb;          // global row (b*T+t)
        const int bb = gr >> 11;
        const int tt = gr & (T_SEQ - 1);
#pragma unroll
        for (int nt = 0; nt < 2; ++nt) {
            const float v = acc[nt][r];
            const int o  = n0 + wn * 32 + nt * 16 + lm;
            const int ol = o & 1023;
            const int hh = ol >> 6;
            const int dd = ol & 63;
            const size_t bh = (size_t)bb * NHEAD + hh;
            if (mat == 0)
                Qb[(bh * T_SEQ + tt) * DHEAD + dd] = (_Float16)(v * 0.125f);
            else if (mat == 1)
                Kb[(bh * T_SEQ + tt) * DHEAD + dd] = (_Float16)v;
            else
                Vt[(bh * DHEAD + dd) * T_SEQ + tt] = (_Float16)v;
        }
    }
}

// ---------------------------------------------------------------------------
// Kernel 2: sliding-window causal attention, flash-style, one wave per
// 16-query tile per (b,h).  Writes O as f16 [b][t][h*64+d].
// ---------------------------------------------------------------------------
__global__ __launch_bounds__(128) void attn_kernel(
    const _Float16* __restrict__ Qb, const _Float16* __restrict__ Kb,
    const _Float16* __restrict__ Vt, _Float16* __restrict__ Ob) {
    __shared__ _Float16 Pb[4][16 * 40];   // per-wave P tile (16 x 32, padded)

    const int tid  = threadIdx.x;
    const int lane = tid & 31;
    const int wv   = tid >> 5;
    const int g    = blockIdx.x * 4 + wv;        // 0..4095
    const int bh   = g >> 7;                     // 0..31
    const int qt   = g & 127;
    const int i0   = qt << 4;
    const int b    = bh >> 4;
    const int h    = bh & 15;

    const _Float16* __restrict__ Q = Qb + (size_t)bh * T_SEQ * DHEAD;
    const _Float16* __restrict__ K = Kb + (size_t)bh * T_SEQ * DHEAD;
    const _Float16* __restrict__ V = Vt + (size_t)bh * DHEAD * T_SEQ;

    const int lm = lane & 15;
    const int hb = (lane < 16) ? 0 : 8;
    const int kb = (lane < 16) ? 0 : 8;
    const int ks = (lane < 16) ? 0 : 16;

    // Q fragments (16 queries x 64 dims, already pre-scaled by 1/8)
    v16h qf[2];
#pragma unroll
    for (int c = 0; c < 2; ++c) {
        const _Float16* qp = Q + (size_t)(i0 + lm) * DHEAD + c * 32;
        qf[c] = ld16(qp + kb, qp + 16 + kb);
    }

    float mrun[8], lrun[8];
    v8f acc[4];
#pragma unroll
    for (int r = 0; r < 8; ++r) { mrun[r] = -1e29f; lrun[r] = 0.0f; }
#pragma unroll
    for (int nt = 0; nt < 4; ++nt) acc[nt] = v8f{};

    int jlo = i0 - HALFWIN;
    if (jlo < 0) jlo = 0;
    const int jstart = jlo & ~31;
    const int npairs = ((i0 + 15 - jstart) >> 5) + 1;

    for (int p = 0; p < npairs; ++p) {
        const int j0 = jstart + p * 32;
        v8f s[2];
#pragma unroll
        for (int t2 = 0; t2 < 2; ++t2) {
            const int jt = j0 + t2 * 16;
            int jc = jt + lm;
            if (jc > T_SEQ - 1) jc = T_SEQ - 1;
            v8f z = v8f{};
#pragma unroll
            for (int c = 0; c < 2; ++c) {
                const _Float16* kp = K + (size_t)jc * DHEAD + c * 32 + kb;
                v16h kf = ld16(kp, kp + 16);
                z = wmma_f16(qf[c], kf, z);
            }
            // mask: 0 <= i-j <= HALFWIN, j < T
#pragma unroll
            for (int r = 0; r < 8; ++r) {
                const int i  = i0 + r + hb;
                const int jj = jt + lm;
                const bool ok = (jj <= i) && (i - jj <= HALFWIN) && (jj < T_SEQ);
                z[r] = ok ? z[r] : -1e30f;
            }
            s[t2] = z;
        }

        // online softmax stats over the 32-key pair
#pragma unroll
        for (int r = 0; r < 8; ++r) {
            float rm = fmaxf(s[0][r], s[1][r]);
            rm = fmaxf(rm, __shfl_xor(rm, 1, 16));
            rm = fmaxf(rm, __shfl_xor(rm, 2, 16));
            rm = fmaxf(rm, __shfl_xor(rm, 4, 16));
            rm = fmaxf(rm, __shfl_xor(rm, 8, 16));
            const float mn    = fmaxf(mrun[r], rm);
            const float alpha = __expf(mrun[r] - mn);
            const float p0 = __expf(s[0][r] - mn);
            const float p1 = __expf(s[1][r] - mn);
            float rs = p0 + p1;
            rs += __shfl_xor(rs, 1, 16);
            rs += __shfl_xor(rs, 2, 16);
            rs += __shfl_xor(rs, 4, 16);
            rs += __shfl_xor(rs, 8, 16);
            lrun[r] = lrun[r] * alpha + rs;
            mrun[r] = mn;
#pragma unroll
            for (int nt = 0; nt < 4; ++nt) acc[nt][r] *= alpha;
            _Float16* pp = &Pb[wv][(r + hb) * 40];
            pp[lm]      = (_Float16)p0;
            pp[16 + lm] = (_Float16)p1;
        }
        __asm__ volatile("s_wait_dscnt 0" ::: "memory");

        // P (16x32) x V^T tiles (32 x 64)
        v16h pf = ld16(&Pb[wv][lm * 40 + kb], &Pb[wv][lm * 40 + 16 + kb]);
#pragma unroll
        for (int nt = 0; nt < 4; ++nt) {
            const _Float16* vp = V + (size_t)(nt * 16 + lm) * T_SEQ + j0 + kb;
            v16h vf = ld16(vp, vp + 16);
            acc[nt] = wmma_f16(pf, vf, acc[nt]);
        }
    }

    // normalize and store O in [b][t][h*64+d] f16 layout
#pragma unroll
    for (int r = 0; r < 8; ++r) {
        const float inv = 1.0f / lrun[r];
        const size_t row = (size_t)(b * T_SEQ + i0 + r + hb);
#pragma unroll
        for (int nt = 0; nt < 4; ++nt) {
            Ob[row * D_MOD + h * DHEAD + nt * 16 + lm] = (_Float16)(acc[nt][r] * inv);
        }
    }
}

// ---------------------------------------------------------------------------
// Kernel 3: output projection  Y = O @ Wo^T   (f16 x f16 -> f32)
// A (f16) asyncs straight into LDS; B (f32) staged f32 + cvt at read.
// ---------------------------------------------------------------------------
__global__ __launch_bounds__(256) void out_gemm_kernel(
    const _Float16* __restrict__ O, const float* __restrict__ Wo,
    float* __restrict__ Y) {
    __shared__ _Float16 As[2][64 * 40];   // f16 tile, stride 40 halfs
    __shared__ float    Bs[2][64 * 44];   // f32 tile, stride 44 floats

    const int m0 = blockIdx.x * 64;
    const int n0 = blockIdx.y * 64;

    const int tid  = threadIdx.x;
    const int lane = tid & 31;
    const int wv   = tid >> 5;
    const int wm   = wv & 3;
    const int wn   = wv >> 2;
    const int lm   = lane & 15;
    const int hb   = (lane < 16) ? 0 : 8;
    const int kb   = (lane < 16) ? 0 : 8;
    const int ks   = (lane < 16) ? 0 : 16;

    const int arow = tid >> 2;
    const int aq   = tid & 3;

    const _Float16* og = O  + (size_t)(m0 + arow) * D_MOD + aq * 8;  // 8 halfs = 16B
    const float*    wg = Wo + (size_t)(n0 + arow) * D_MOD + aq * 8;  // 8 floats = 2x16B
    const uint32_t ldsA0 = (uint32_t)(size_t)&As[0][arow * 40 + aq * 8];
    const uint32_t ldsA1 = (uint32_t)(size_t)&As[1][arow * 40 + aq * 8];
    const uint32_t ldsB0 = (uint32_t)(size_t)&Bs[0][arow * 44 + aq * 8];
    const uint32_t ldsB1 = (uint32_t)(size_t)&Bs[1][arow * 44 + aq * 8];

    v8f acc[2];
#pragma unroll
    for (int nt = 0; nt < 2; ++nt) acc[nt] = v8f{};

    async_b128(ldsA0, og);
    async_b128(ldsB0, wg);      async_b128(ldsB0 + 16, wg + 4);

    for (int it = 0; it < 32; ++it) {
        const int cur = it & 1;
        wait_async0();
        __syncthreads();
        if (it + 1 < 32) {
            const int kn = (it + 1) * 32;
            const uint32_t la = cur ? ldsA0 : ldsA1;
            const uint32_t lb = cur ? ldsB0 : ldsB1;
            async_b128(la, og + kn);
            async_b128(lb, wg + kn);  async_b128(lb + 16, wg + kn + 4);
        }
        const int mloc = wm * 16 + lm;
        v16h af = ld16(&As[cur][mloc * 40 + kb], &As[cur][mloc * 40 + 16 + kb]);
#pragma unroll
        for (int nt = 0; nt < 2; ++nt) {
            const int nloc = wn * 32 + nt * 16 + lm;
            v16h bf = cvt16(&Bs[cur][nloc * 44 + kb], &Bs[cur][nloc * 44 + 16 + kb]);
            acc[nt] = wmma_f16(af, bf, acc[nt]);
        }
    }

#pragma unroll
    for (int r = 0; r < 8; ++r) {
        const int gr = m0 + wm * 16 + r + hb;
#pragma unroll
        for (int nt = 0; nt < 2; ++nt) {
            const int o = n0 + wn * 32 + nt * 16 + lm;
            Y[(size_t)gr * D_MOD + o] = acc[nt][r];
        }
    }
}

// ---------------------------------------------------------------------------
extern "C" void kernel_launch(void* const* d_in, const int* in_sizes, int n_in,
                              void* d_out, int out_size, void* d_ws, size_t ws_size,
                              hipStream_t stream) {
    (void)in_sizes; (void)n_in; (void)out_size; (void)ws_size;
    const float* x  = (const float*)d_in[0];
    const float* Wq = (const float*)d_in[1];
    const float* Wk = (const float*)d_in[2];
    const float* Wv = (const float*)d_in[3];
    const float* Wo = (const float*)d_in[4];
    float* Y = (float*)d_out;

    char* ws = (char*)d_ws;
    const size_t MB = 1024ull * 1024ull;
    _Float16* Qb = (_Float16*)(ws + 0 * MB);   // 8 MB: [bh][T][dh]
    _Float16* Kb = (_Float16*)(ws + 8 * MB);   // 8 MB: [bh][T][dh]
    _Float16* Vt = (_Float16*)(ws + 16 * MB);  // 8 MB: [bh][dh][T]
    _Float16* Ob = (_Float16*)(ws + 24 * MB);  // 8 MB: [b*T][D]

    qkv_gemm_kernel<<<dim3(64, 48), 256, 0, stream>>>(x, Wq, Wk, Wv, Qb, Kb, Vt);
    attn_kernel<<<dim3(1024), 128, 0, stream>>>(Qb, Kb, Vt, Ob);
    out_gemm_kernel<<<dim3(64, 16), 256, 0, stream>>>(Ob, Wo, Y);
}
